// MultiHeadAttention_80118319940348
// MI455X (gfx1250) — matrix operand order repaired and checked by reading, not compiled, not run on hardware
//
#include <hip/hip_runtime.h>
#include <hip/hip_bf16.h>
#include <math.h>

// ---------------------------------------------------------------------------
// Problem constants (fixed by setup_inputs)
// ---------------------------------------------------------------------------
#define BDIM   2
#define HDIM   16
#define SDIM   2048
#define KVDIM  4096          // C + S
#define DH     64
#define HID    1024          // H * Dh
#define MROWS  (BDIM * SDIM) // 4096 rows into the projection GEMM

typedef __attribute__((ext_vector_type(16))) __bf16 v16bf;
typedef __attribute__((ext_vector_type(8)))  float  v8f;
typedef __attribute__((ext_vector_type(4)))  float  f4;

#define WMMA_BF16(a, b, c) \
  __builtin_amdgcn_wmma_f32_16x16x32_bf16(false, (a), false, (b), (short)0, (c), false, false)

// ---------------------------------------------------------------------------
// WMMA operand loaders (A/B in this code's contiguous k order; C/D per §7.12.2)
//
// A (16x32 bf16): lane holds row M = lane&15.  Element e maps to
//   K = e + (lane>=16 ? 16 : 0)
// B (32x16 bf16): lane holds col N = lane&15.  Element e maps to
//   K = e + (lane>=16 ? 16 : 0)      (caller bakes the +16 into the pointer)
// C/D (16x16 f32): VGPR r, lanes 0-15 -> M=r, N=lane;
//                  lanes 16-31 -> M=r+8, N=lane-16
// ---------------------------------------------------------------------------
__device__ __forceinline__ v16bf load_A_f32(const float* rowPtr, int kBase, int half8) {
  v16bf a;
#pragma unroll
  for (int e = 0; e < 16; ++e) {
    int kk = kBase + e + 2 * half8;
    a[e] = (__bf16)rowPtr[kk];
  }
  return a;
}

__device__ __forceinline__ v16bf load_B_contig(const float* p) {
  v16bf b;
#pragma unroll
  for (int e = 0; e < 16; ++e) b[e] = (__bf16)p[e];
  return b;
}

__device__ __forceinline__ v16bf load_B_strided(const float* p, int stride) {
  v16bf b;
#pragma unroll
  for (int e = 0; e < 16; ++e) b[e] = (__bf16)p[(size_t)e * (size_t)stride];
  return b;
}

__device__ __forceinline__ v16bf load_A_lds(const __bf16* Ps, int row, int half8) {
  v16bf a;
#pragma unroll
  for (int e = 0; e < 16; ++e) {
    int kk = e + 2 * half8;
    a[e] = Ps[row * 32 + kk];
  }
  return a;
}

// Row reductions over the 16 columns of a score tile: each lane half
// (0-15 / 16-31) owns a disjoint row set, xor masks 1..8 stay inside a half.
__device__ __forceinline__ float rowmax16(float x) {
#pragma unroll
  for (int m = 1; m <= 8; m <<= 1) x = fmaxf(x, __shfl_xor(x, m, 32));
  return x;
}
__device__ __forceinline__ float rowsum16(float x) {
#pragma unroll
  for (int m = 1; m <= 8; m <<= 1) x += __shfl_xor(x, m, 32);
  return x;
}

// ---------------------------------------------------------------------------
// Kernel 1: flash attention.  One wave per (b, h, 16-row query tile).
// Writes pre-projection attn in [B, S, H*Dh] layout (fp32) to workspace.
// ---------------------------------------------------------------------------
__global__ __launch_bounds__(32)
void attn_flash_kernel(const float* __restrict__ q, const float* __restrict__ k,
                       const float* __restrict__ v, const int* __restrict__ memlen_p,
                       float* __restrict__ attn_ws) {
  const int memL = memlen_p[0];
  const int restrictedCol = (KVDIM - SDIM) + memL;   // cols >= this are restricted

  const int tilesPerHead = SDIM / 16;                // 128
  int wid   = blockIdx.x;
  int mTile = wid % tilesPerHead;
  int bh    = wid / tilesPerHead;
  int h     = bh % HDIM;
  int bb    = bh / HDIM;
  int rowBase = mTile * 16;

  int lane  = threadIdx.x;
  int col   = lane & 15;
  int half8 = (lane >> 4) << 3;        // 0 for lanes 0-15, 8 for lanes 16-31
  int koff16 = (lane >= 16) ? 16 : 0;

  const float* Q = q + ((size_t)(bb * HDIM + h) * SDIM) * DH;
  const float* K = k + ((size_t)(bb * HDIM + h) * KVDIM) * DH;
  const float* V = v + ((size_t)(bb * HDIM + h) * KVDIM) * DH;

  // Q tile as two A operands (Dh=64 -> K-steps 0..31 and 32..63)
  const float* qrow = Q + (size_t)(rowBase + col) * DH;
  v16bf aq0 = load_A_f32(qrow, 0,  half8);
  v16bf aq1 = load_A_f32(qrow, 32, half8);

  v8f o0 = {}, o1 = {}, o2 = {}, o3 = {};
  float m[8], l[8];
#pragma unroll
  for (int r = 0; r < 8; ++r) { m[r] = -INFINITY; l[r] = 0.0f; }

  __shared__ __bf16 Ps[16 * 32];       // P tile, row-major [row][key]

  const float scale = 0.125f;          // 1/sqrt(64)

  for (int kb = 0; kb < KVDIM; kb += 32) {
    // ---- scores: two 16-key tiles, each accumulated over Dh with 2 WMMAs
    v8f s0 = {}, s1 = {};
    {
      const float* kr0 = K + (size_t)(kb + col) * DH + koff16;
      s0 = WMMA_BF16(aq0, load_B_contig(kr0),      s0);
      s0 = WMMA_BF16(aq1, load_B_contig(kr0 + 32), s0);
      const float* kr1 = K + (size_t)(kb + 16 + col) * DH + koff16;
      s1 = WMMA_BF16(aq0, load_B_contig(kr1),      s1);
      s1 = WMMA_BF16(aq1, load_B_contig(kr1 + 32), s1);
    }

    // ---- scale, mask, online softmax (8 rows per lane)
    float p0[8], p1[8];
#pragma unroll
    for (int r = 0; r < 8; ++r) {
      int row = rowBase + r + half8;
      int c0  = kb + col;
      int c1  = kb + 16 + col;
      float x0 = s0[r] * scale;
      float x1 = s1[r] * scale;
      bool mk0 = (c0 >= restrictedCol) && ((row < memL) || ((c0 - restrictedCol) > (row - memL)));
      bool mk1 = (c1 >= restrictedCol) && ((row < memL) || ((c1 - restrictedCol) > (row - memL)));
      if (mk0) x0 = -INFINITY;
      if (mk1) x1 = -INFINITY;

      float cmax = rowmax16(fmaxf(x0, x1));
      float mn   = fmaxf(m[r], cmax);
      float alpha = (m[r] == -INFINITY) ? 0.0f : __expf(m[r] - mn);
      p0[r] = __expf(x0 - mn);
      p1[r] = __expf(x1 - mn);
      float csum = rowsum16(p0[r] + p1[r]);
      l[r] = l[r] * alpha + csum;
      m[r] = mn;
      o0[r] *= alpha; o1[r] *= alpha; o2[r] *= alpha; o3[r] *= alpha;
    }

    // ---- re-layout P (C/D format -> A format) through LDS
    __syncthreads();
#pragma unroll
    for (int r = 0; r < 8; ++r) {
      Ps[(r + half8) * 32 + col]      = (__bf16)p0[r];
      Ps[(r + half8) * 32 + 16 + col] = (__bf16)p1[r];
    }
    __syncthreads();
    v16bf ap = load_A_lds(Ps, col, half8);

    // ---- P x V : 4 N-tiles of 16, K=32 keys
    const float* vp = V + (size_t)(kb + koff16) * DH + col;
    o0 = WMMA_BF16(ap, load_B_strided(vp,      DH), o0);
    o1 = WMMA_BF16(ap, load_B_strided(vp + 16, DH), o1);
    o2 = WMMA_BF16(ap, load_B_strided(vp + 32, DH), o2);
    o3 = WMMA_BF16(ap, load_B_strided(vp + 48, DH), o3);
  }

  // ---- epilogue: normalize, write attn in [B, S, H*Dh] layout
#pragma unroll
  for (int r = 0; r < 8; ++r) {
    float inv = 1.0f / l[r];
    int row = rowBase + r + half8;
    float* dst = attn_ws + ((size_t)(bb * SDIM + row) * HDIM + h) * DH + col;
    dst[0]  = o0[r] * inv;
    dst[16] = o1[r] * inv;
    dst[32] = o2[r] * inv;
    dst[48] = o3[r] * inv;
  }
}

// ---------------------------------------------------------------------------
// Kernel 2: output projection  out = attn @ W^T + b
// One wave per 16x64 output tile, K = 1024 loop in steps of 32.
// ---------------------------------------------------------------------------
__global__ __launch_bounds__(32)
void proj_kernel(const float* __restrict__ attn_ws, const float* __restrict__ W,
                 const float* __restrict__ bias, float* __restrict__ out) {
  const int nTiles = HID / 64;                 // 16
  int wid   = blockIdx.x;
  int nBase = (wid % nTiles) * 64;
  int mBase = (wid / nTiles) * 16;

  int lane  = threadIdx.x;
  int col   = lane & 15;
  int half8 = (lane >> 4) << 3;
  int koff16 = (lane >= 16) ? 16 : 0;

  v8f c0 = {}, c1 = {}, c2 = {}, c3 = {};
  const float* arow = attn_ws + (size_t)(mBase + col) * HID;

  for (int kb = 0; kb < HID; kb += 32) {
    v16bf a = load_A_f32(arow, kb, half8);
    // B[kk][n] = W[n][kb + kk]  (out = attn @ W^T), contiguous in k
    const float* w0 = W + (size_t)(nBase +  0 + col) * HID + kb + koff16;
    const float* w1 = W + (size_t)(nBase + 16 + col) * HID + kb + koff16;
    const float* w2 = W + (size_t)(nBase + 32 + col) * HID + kb + koff16;
    const float* w3 = W + (size_t)(nBase + 48 + col) * HID + kb + koff16;
    c0 = WMMA_BF16(a, load_B_contig(w0), c0);
    c1 = WMMA_BF16(a, load_B_contig(w1), c1);
    c2 = WMMA_BF16(a, load_B_contig(w2), c2);
    c3 = WMMA_BF16(a, load_B_contig(w3), c3);
  }

#pragma unroll
  for (int r = 0; r < 8; ++r) {
    int mrow = mBase + r + half8;
    float* dst = out + (size_t)mrow * HID + nBase + col;
    dst[0]  = c0[r] + bias[nBase +  0 + col];
    dst[16] = c1[r] + bias[nBase + 16 + col];
    dst[32] = c2[r] + bias[nBase + 32 + col];
    dst[48] = c3[r] + bias[nBase + 48 + col];
  }
}

// ---------------------------------------------------------------------------
// Kernel 3: stream out the full float mask [B,H,S,KV] (1.07 GB).
// Non-temporal float4 stores: write-once stream, keep it out of L2.
// ---------------------------------------------------------------------------
__global__ void mask_kernel(const int* __restrict__ memlen_p, float* __restrict__ maskOut) {
  const int memL = memlen_p[0];
  const int restrictedCol = (KVDIM - SDIM) + memL;
  const size_t total4 = (size_t)BDIM * HDIM * SDIM * KVDIM / 4;

  size_t stride = (size_t)gridDim.x * blockDim.x;
  for (size_t idx = (size_t)blockIdx.x * blockDim.x + threadIdx.x; idx < total4; idx += stride) {
    size_t e = idx * 4;                       // 4 consecutive cols, same row
    int t = (int)(e % KVDIM);
    int s = (int)((e / KVDIM) % SDIM);
    f4 val;
#pragma unroll
    for (int j = 0; j < 4; ++j) {
      int tt = t + j;
      bool mk = (tt >= restrictedCol) &&
                ((s < memL) || ((tt - restrictedCol) > (s - memL)));
      val[j] = mk ? 1.0f : 0.0f;
    }
    __builtin_nontemporal_store(val, ((f4*)maskOut) + idx);
  }
}

// ---------------------------------------------------------------------------
// Launch
// ---------------------------------------------------------------------------
extern "C" void kernel_launch(void* const* d_in, const int* in_sizes, int n_in,
                              void* d_out, int out_size, void* d_ws, size_t ws_size,
                              hipStream_t stream) {
  (void)in_sizes; (void)n_in; (void)out_size; (void)ws_size;

  const float* q      = (const float*)d_in[0];
  const float* k      = (const float*)d_in[1];
  const float* v      = (const float*)d_in[2];
  const int*   memlen = (const int*)  d_in[3];
  const float* W      = (const float*)d_in[4];
  const float* bias   = (const float*)d_in[5];

  float* out      = (float*)d_out;
  float* mask_out = out + (size_t)BDIM * SDIM * HID;   // tuple: (out, mask_full)
  float* attn_ws  = (float*)d_ws;                      // 2*2048*1024 fp32 = 16.8 MB

  // 1) flash attention: 2*16*128 = 4096 waves
  attn_flash_kernel<<<BDIM * HDIM * (SDIM / 16), 32, 0, stream>>>(q, k, v, memlen, attn_ws);

  // 2) projection GEMM: (4096/16) * (1024/64) = 4096 waves
  proj_kernel<<<(MROWS / 16) * (HID / 64), 32, 0, stream>>>(attn_ws, W, bias, out);

  // 3) mask stream (independent; same stream serializes it harmlessly)
  mask_kernel<<<65536, 256, 0, stream>>>(memlen, mask_out);
}
